// SelfAttentionModel_9577777070307
// MI455X (gfx1250) — matrix operand order repaired and checked by reading, not compiled, not run on hardware
//
#include <hip/hip_runtime.h>
#include <hip/hip_bf16.h>
#include <math.h>

// ---------------------------------------------------------------------------
// Problem constants (from the reference)
// ---------------------------------------------------------------------------
#define BB        64
#define SSEQ      512
#define FF        150
#define HIDN      100
#define DD        401          // HID + 2F + 1
#define DP        416          // DD padded to multiple of 32 (13*32)
#define STAT      64
#define FCIN      465          // DD + STAT
#define FCHID     250

typedef __bf16 bf16_t;
typedef __bf16 v16bf __attribute__((ext_vector_type(16)));
typedef __bf16 v8bf  __attribute__((ext_vector_type(8)));
typedef float  v8f   __attribute__((ext_vector_type(8)));

// ---------------------------------------------------------------------------
// Core tiled GEMM on WMMA bf16 (f32 accumulate).
//   C[b] = A[b] (MxK, row major, bf16) * B[b] (KxN)  [+ bias[n]]
// B is supplied TRANSPOSED: BT (NxK row major, bf16) so every fragment load
// is a contiguous 16B/32B vector load matching the ISA operand layouts.
// One wave32 -> one 32x32 macro-tile (2x2 WMMA tiles): A/B fragments are each
// reused twice, cutting VMEM issue per WMMA from 4 loads to 2.
// 8 waves / 256-thread block.  Requires M,N multiples of 32, K multiple of 32.
// ---------------------------------------------------------------------------
__global__ __launch_bounds__(256)
void gemm_bf16_wmma(const bf16_t* __restrict__ A, long long sA, int lda,
                    const bf16_t* __restrict__ BT, long long sB, int ldb,
                    void* __restrict__ Cout, long long sC, int ldc,
                    const float* __restrict__ bias,
                    int M, int N, int K, int nbatch, int f32out, int transC)
{
    const int mt = M >> 5, nt = N >> 5;             // 32x32 macro tiles
    const long long perBatch = (long long)mt * nt;
    const long long tile = (long long)blockIdx.x * 8 + (threadIdx.x >> 5);
    if (tile >= perBatch * nbatch) return;          // wave-uniform exit: EXEC all-1

    const int b  = (int)(tile / perBatch);
    const int t  = (int)(tile - (long long)b * perBatch);
    const int tm = (t / nt) << 5;                   // macro-tile row origin
    const int tn = (t - (t / nt) * nt) << 5;        // macro-tile col origin

    const int lane = threadIdx.x & 31;
    const int r    = lane & 15;      // A row / B column / C column within tile
    const int hi   = lane >> 4;      // lane half selects K-subblock

    const bf16_t* Arow0 = A  + (long long)b * sA + (long long)(tm + r) * lda;
    const bf16_t* Arow1 = Arow0 + (long long)16 * lda;
    const bf16_t* Brow0 = BT + (long long)b * sB + (long long)(tn + r) * ldb;
    const bf16_t* Brow1 = Brow0 + (long long)16 * ldb;

    v8f acc00 = {0.f,0.f,0.f,0.f,0.f,0.f,0.f,0.f};
    v8f acc01 = acc00, acc10 = acc00, acc11 = acc00;

    for (int k0 = 0; k0 < K; k0 += 32) {
        // speculative prefetch of the K block two ahead (dropped if OOB)
        __builtin_prefetch(Arow0 + k0 + 64, 0, 1);
        __builtin_prefetch(Arow1 + k0 + 64, 0, 1);
        __builtin_prefetch(Brow0 + k0 + 64, 0, 1);
        __builtin_prefetch(Brow1 + k0 + 64, 0, 1);

        // A 16x32: lanes 0-15 hold K 0..7 & 16..23, lanes 16-31 hold K 8..15 & 24..31
        v8bf a0lo = *(const v8bf*)(Arow0 + k0 + hi * 8);
        v8bf a0hi = *(const v8bf*)(Arow0 + k0 + 16 + hi * 8);
        v16bf a0 = __builtin_shufflevector(a0lo, a0hi,
                       0,1,2,3,4,5,6,7,8,9,10,11,12,13,14,15);
        v8bf a1lo = *(const v8bf*)(Arow1 + k0 + hi * 8);
        v8bf a1hi = *(const v8bf*)(Arow1 + k0 + 16 + hi * 8);
        v16bf a1 = __builtin_shufflevector(a1lo, a1hi,
                       0,1,2,3,4,5,6,7,8,9,10,11,12,13,14,15);
        // B 32x16: lane = column n, lanes 0-15 hold K 0..7 & 16..23, lanes 16-31 hold K 8..15 & 24..31
        v8bf b0lo = *(const v8bf*)(Brow0 + k0 + hi * 8);
        v8bf b0hi = *(const v8bf*)(Brow0 + k0 + 16 + hi * 8);
        v16bf b0 = __builtin_shufflevector(b0lo, b0hi,
                       0,1,2,3,4,5,6,7,8,9,10,11,12,13,14,15);
        v8bf b1lo = *(const v8bf*)(Brow1 + k0 + hi * 8);
        v8bf b1hi = *(const v8bf*)(Brow1 + k0 + 16 + hi * 8);
        v16bf b1 = __builtin_shufflevector(b1lo, b1hi,
                       0,1,2,3,4,5,6,7,8,9,10,11,12,13,14,15);

        acc00 = __builtin_amdgcn_wmma_f32_16x16x32_bf16(false, a0, false, b0, (short)0, acc00, false, false);
        acc01 = __builtin_amdgcn_wmma_f32_16x16x32_bf16(false, a0, false, b1, (short)0, acc01, false, false);
        acc10 = __builtin_amdgcn_wmma_f32_16x16x32_bf16(false, a1, false, b0, (short)0, acc10, false, false);
        acc11 = __builtin_amdgcn_wmma_f32_16x16x32_bf16(false, a1, false, b1, (short)0, acc11, false, false);
    }

    // store 4 tiles; C layout: m = hi*8 + vgpr index, n = lane&15
    const long long cbase = (long long)b * sC;
#pragma unroll
    for (int q = 0; q < 4; ++q) {
        const v8f acc = (q == 0) ? acc00 : (q == 1) ? acc01 : (q == 2) ? acc10 : acc11;
        const int gm0 = tm + ((q >> 1) << 4);
        const int gn  = tn + ((q & 1) << 4) + r;
        const float bv = bias ? bias[gn] : 0.f;
#pragma unroll
        for (int v = 0; v < 8; ++v) {
            const int gm = gm0 + hi * 8 + v;
            const float val = acc[v] + bv;
            const long long idx = transC
                ? cbase + (long long)gn * ldc + gm
                : cbase + (long long)gm * ldc + gn;
            if (f32out) ((float*)Cout)[idx] = val;
            else        ((bf16_t*)Cout)[idx] = (bf16_t)val;
        }
    }
}

// ---------------------------------------------------------------------------
// Build x = concat(seq, ts, emb[codes], tsc) -> bf16 [B,S,DP], zero padded.
// ---------------------------------------------------------------------------
__global__ __launch_bounds__(128)
void pack_x(const float* __restrict__ seqs, const float* __restrict__ ts,
            const int* __restrict__ codes, const float* __restrict__ tsc,
            const float* __restrict__ emb, bf16_t* __restrict__ xb)
{
    const int bs   = blockIdx.x;               // b*SSEQ + s
    const int code = codes[bs];
    const long long obase = (long long)bs * DP;
    for (int d = threadIdx.x; d < DP; d += 128) {
        float v;
        if      (d < FF)            v = seqs[(long long)bs * FF + d];
        else if (d < 2 * FF)        v = ts[(long long)bs * FF + (d - FF)];
        else if (d < 2 * FF + HIDN) v = emb[(long long)code * HIDN + (d - 2 * FF)];
        else if (d == DD - 1)       v = tsc[bs];
        else                        v = 0.f;
        xb[obase + d] = (bf16_t)v;
    }
}

// W [K,N] f32 -> WT [Np,Kp] bf16 (transposed, zero padded)
__global__ void prep_w(const float* __restrict__ W, bf16_t* __restrict__ WT,
                       int K, int N, int Kp, int Np)
{
    const int k = blockIdx.x * 16 + threadIdx.x;
    const int n = blockIdx.y * 16 + threadIdx.y;
    if (k < Kp && n < Np)
        WT[(long long)n * Kp + k] =
            (bf16_t)((k < K && n < N) ? W[(long long)k * N + n] : 0.f);
}

__global__ void pad_bias(const float* __restrict__ src, float* __restrict__ dst,
                         int N, int Np)
{
    const int i = blockIdx.x * blockDim.x + threadIdx.x;
    if (i < Np) dst[i] = (i < N) ? src[i] : 0.f;
}

// ---------------------------------------------------------------------------
// Row softmax with scale: reads f32 scores, writes f32 probs (unpadded, to
// d_out) and bf16 probs (padded, zero-filled) for the next WMMA GEMM.
// One 256-thread block per row.
// ---------------------------------------------------------------------------
__global__ __launch_bounds__(256)
void softmax_rows(const float* __restrict__ Sin, float* __restrict__ Pf32,
                  bf16_t* __restrict__ Pbf,
                  int rows_in, int rows_valid, int cols_pad, int cols_valid,
                  float scale)
{
    __shared__ float red[256];
    const int tid = threadIdx.x;
    const int bid = blockIdx.x;
    const int b = bid / rows_in;
    const int r = bid - b * rows_in;
    const long long ibase = ((long long)b * rows_in + r) * cols_pad;

    if (r >= rows_valid) {                       // padded row: zero bf16 probs
        for (int c = tid; c < cols_pad; c += 256) Pbf[ibase + c] = (bf16_t)0.f;
        return;
    }
    float m = -3.4e38f;
    for (int c = tid; c < cols_valid; c += 256)
        m = fmaxf(m, Sin[ibase + c] * scale);
    red[tid] = m; __syncthreads();
    for (int off = 128; off > 0; off >>= 1) {
        if (tid < off) red[tid] = fmaxf(red[tid], red[tid + off]);
        __syncthreads();
    }
    m = red[0]; __syncthreads();

    float sum = 0.f;
    for (int c = tid; c < cols_valid; c += 256)
        sum += __expf(Sin[ibase + c] * scale - m);
    red[tid] = sum; __syncthreads();
    for (int off = 128; off > 0; off >>= 1) {
        if (tid < off) red[tid] += red[tid + off];
        __syncthreads();
    }
    const float inv = 1.f / red[0];

    const long long obase = ((long long)b * rows_valid + r) * cols_valid;
    for (int c = tid; c < cols_valid; c += 256) {
        const float p = __expf(Sin[ibase + c] * scale - m) * inv;
        Pf32[obase + c] = p;
        Pbf[ibase + c]  = (bf16_t)p;
    }
    for (int c = cols_valid + tid; c < cols_pad; c += 256)
        Pbf[ibase + c] = (bf16_t)0.f;
}

// y[b,d,s] = av1[b,s,d] + PE(s,d)  (d<DD), zero for padded d rows.
__global__ __launch_bounds__(256)
void build_y(const bf16_t* __restrict__ av1, bf16_t* __restrict__ y)
{
    const int bid = blockIdx.x;                  // b*DP + d
    const int b = bid / DP, d = bid - b * DP;
    const long long ybase = ((long long)b * DP + d) * SSEQ;
    if (d >= DD) {
        for (int s = threadIdx.x; s < SSEQ; s += 256) y[ybase + s] = (bf16_t)0.f;
        return;
    }
    const float expo = (float)(2 * (d >> 1)) / (float)DD;
    const float rate = __expf(-expo * 9.210340371976184f);   // 10000^-expo
    const bool  iscos = d & 1;
    for (int s = threadIdx.x; s < SSEQ; s += 256) {
        const float pe = iscos ? __cosf(s * rate) : __sinf(s * rate);
        const float av = (float)av1[((long long)b * SSEQ + s) * DP + d];
        y[ybase + s] = (bf16_t)(av + pe);
    }
}

// agg[b,d] = mean_s av2[b,d,s] -> h[b, d]
__global__ __launch_bounds__(256)
void mean_rows(const float* __restrict__ av2, float* __restrict__ h)
{
    __shared__ float red[256];
    const int bid = blockIdx.x;                  // b*DD + d
    const int b = bid / DD, d = bid - b * DD;
    const long long base = ((long long)b * DP + d) * SSEQ;
    float s = 0.f;
    for (int i = threadIdx.x; i < SSEQ; i += 256) s += av2[base + i];
    red[threadIdx.x] = s; __syncthreads();
    for (int off = 128; off > 0; off >>= 1) {
        if (threadIdx.x < off) red[threadIdx.x] += red[threadIdx.x + off];
        __syncthreads();
    }
    if (threadIdx.x == 0)
        h[(long long)b * FCIN + d] = red[0] * (1.f / (float)SSEQ);
}

// Static attention over batch dim: 64x64 everything, one block, LDS resident.
__global__ __launch_bounds__(256)
void static_attn(const float* __restrict__ sv,
                 const float* __restrict__ qw, const float* __restrict__ qb,
                 const float* __restrict__ kw, const float* __restrict__ kb,
                 const float* __restrict__ vw, const float* __restrict__ vb,
                 float* __restrict__ aw3, float* __restrict__ h)
{
    __shared__ float Q[64 * 64], Ks[64 * 64], A[64 * 64];
    const int tid = threadIdx.x;
    for (int idx = tid; idx < 64 * 64; idx += 256) {
        const int i = idx >> 6, j = idx & 63;
        float aq = qb[j], ak = kb[j];
        for (int k = 0; k < 64; ++k) {
            const float s = sv[i * 64 + k];
            aq += s * qw[k * 64 + j];
            ak += s * kw[k * 64 + j];
        }
        Q[idx] = aq; Ks[idx] = ak;
    }
    __syncthreads();
    for (int idx = tid; idx < 64 * 64; idx += 256) {
        const int i = idx >> 6, j = idx & 63;
        float a = 0.f;
        for (int k = 0; k < 64; ++k) a += Q[i * 64 + k] * Ks[j * 64 + k];
        A[idx] = a * 0.125f;                      // 1/sqrt(64)
    }
    __syncthreads();
    if (tid < 64) {
        float m = -3.4e38f;
        for (int j = 0; j < 64; ++j) m = fmaxf(m, A[tid * 64 + j]);
        float s = 0.f;
        for (int j = 0; j < 64; ++j) { const float e = __expf(A[tid * 64 + j] - m); A[tid * 64 + j] = e; s += e; }
        const float inv = 1.f / s;
        for (int j = 0; j < 64; ++j) A[tid * 64 + j] *= inv;
    }
    __syncthreads();
    for (int idx = tid; idx < 64 * 64; idx += 256) aw3[idx] = A[idx];
    for (int idx = tid; idx < 64 * 64; idx += 256) {   // V into Ks
        const int i = idx >> 6, j = idx & 63;
        float av = vb[j];
        for (int k = 0; k < 64; ++k) av += sv[i * 64 + k] * vw[k * 64 + j];
        Ks[idx] = av;
    }
    __syncthreads();
    for (int idx = tid; idx < 64 * 64; idx += 256) {
        const int i = idx >> 6, j = idx & 63;
        float a = 0.f;
        for (int k = 0; k < 64; ++k) a += A[i * 64 + k] * Ks[k * 64 + j];
        h[(long long)i * FCIN + DD + j] = a;
    }
}

// out = sigmoid(relu(h @ fc1 + b1) @ fc2 + b2); one block, hidden in LDS.
__global__ __launch_bounds__(256)
void mlp_head(const float* __restrict__ h,
              const float* __restrict__ w1, const float* __restrict__ b1,
              const float* __restrict__ w2, const float* __restrict__ b2,
              float* __restrict__ out)
{
    __shared__ float hid[64 * FCHID];
    const int tid = threadIdx.x;
    for (int idx = tid; idx < 64 * FCHID; idx += 256) {
        const int i = idx / FCHID, j = idx - i * FCHID;
        float a = b1[j];
        for (int k = 0; k < FCIN; ++k) a += h[i * FCIN + k] * w1[k * FCHID + j];
        hid[idx] = fmaxf(a, 0.f);
    }
    __syncthreads();
    if (tid < 64) {
        float a = b2[0];
        for (int k = 0; k < FCHID; ++k) a += hid[tid * FCHID + k] * w2[k];
        out[tid] = 1.f / (1.f + __expf(-a));
    }
}

// ---------------------------------------------------------------------------
// Host-side orchestration
// ---------------------------------------------------------------------------
extern "C" void kernel_launch(void* const* d_in, const int* in_sizes, int n_in,
                              void* d_out, int out_size, void* d_ws, size_t ws_size,
                              hipStream_t stream)
{
    (void)in_sizes; (void)n_in; (void)out_size; (void)ws_size;

    const float* tsc   = (const float*)d_in[0];
    const int*   codes = (const int*)d_in[1];
    const float* seqs  = (const float*)d_in[2];
    const float* ts    = (const float*)d_in[3];
    const float* sv    = (const float*)d_in[4];
    const float* emb   = (const float*)d_in[5];
    const float* faq_w = (const float*)d_in[6],  *faq_b = (const float*)d_in[7];
    const float* fak_w = (const float*)d_in[8],  *fak_b = (const float*)d_in[9];
    const float* fav_w = (const float*)d_in[10], *fav_b = (const float*)d_in[11];
    const float* taq_w = (const float*)d_in[12], *taq_b = (const float*)d_in[13];
    const float* tak_w = (const float*)d_in[14], *tak_b = (const float*)d_in[15];
    const float* tav_w = (const float*)d_in[16], *tav_b = (const float*)d_in[17];
    const float* saq_w = (const float*)d_in[18], *saq_b = (const float*)d_in[19];
    const float* sak_w = (const float*)d_in[20], *sak_b = (const float*)d_in[21];
    const float* sav_w = (const float*)d_in[22], *sav_b = (const float*)d_in[23];
    const float* fc1_w = (const float*)d_in[24], *fc1_b = (const float*)d_in[25];
    const float* fc2_w = (const float*)d_in[26], *fc2_b = (const float*)d_in[27];

    // d_out layout: out[64] | aw1[64*512*512] | aw2[64*401*401] | aw3[64*64]
    float* out_main = (float*)d_out;
    float* out_aw1  = out_main + 64;
    float* out_aw2  = out_aw1 + (size_t)BB * SSEQ * SSEQ;
    float* out_aw3  = out_aw2 + (size_t)BB * DD * DD;

    // workspace bump allocator (~213 MB total, heavy buffer reuse)
    char* ws = (char*)d_ws;
    size_t off = 0;
    auto bump = [&](size_t bytes) {
        size_t r = off; off = (off + bytes + 255) & ~(size_t)255; return r;
    };
    const size_t SZ_BSDP = (size_t)BB * SSEQ * DP * sizeof(bf16_t);
    const size_t o_xb  = bump(SZ_BSDP);                            // x bf16, later q2
    const size_t o_q1  = bump(SZ_BSDP);                            // q1, later av1, later v2T
    const size_t o_k1  = bump(SZ_BSDP);                            // k1, later y
    const size_t o_v1t = bump(SZ_BSDP);                            // v1^T, later k2
    const size_t o_s   = bump((size_t)BB * SSEQ * SSEQ * 4);       // scores f32 (s1/s2/av2)
    const size_t o_p   = bump((size_t)BB * SSEQ * SSEQ * 2);       // probs bf16 (p1/p2)
    const size_t o_faqT = bump((size_t)DP * DP * 2);
    const size_t o_fakT = bump((size_t)DP * DP * 2);
    const size_t o_favT = bump((size_t)DP * DP * 2);
    const size_t o_taqT = bump((size_t)SSEQ * SSEQ * 2);
    const size_t o_takT = bump((size_t)SSEQ * SSEQ * 2);
    const size_t o_tavT = bump((size_t)SSEQ * SSEQ * 2);
    const size_t o_bfaq = bump(DP * 4), o_bfak = bump(DP * 4), o_bfav = bump(DP * 4);
    const size_t o_btaq = bump(SSEQ * 4), o_btak = bump(SSEQ * 4), o_btav = bump(SSEQ * 4);
    const size_t o_h    = bump((size_t)BB * FCIN * 4);

    auto BF = [&](size_t o) { return (bf16_t*)(ws + o); };
    auto FP = [&](size_t o) { return (float*)(ws + o); };

    auto gemm = [&](const bf16_t* A, long long sA, int lda,
                    const bf16_t* BT, long long sB, int ldb,
                    void* C, long long sC, int ldc, const float* bias,
                    int M, int N, int K, int f32out, int transC) {
        long long tiles = (long long)(M / 32) * (N / 32) * BB;
        int blocks = (int)((tiles + 7) / 8);
        gemm_bf16_wmma<<<blocks, 256, 0, stream>>>(A, sA, lda, BT, sB, ldb,
                                                   C, sC, ldc, bias,
                                                   M, N, K, BB, f32out, transC);
    };
    auto prep = [&](const float* W, bf16_t* WT, int K, int N, int Kp, int Np) {
        dim3 g(Kp / 16, Np / 16), blk(16, 16);
        prep_w<<<g, blk, 0, stream>>>(W, WT, K, N, Kp, Np);
    };
    auto padb = [&](const float* b, float* dst, int N, int Np) {
        pad_bias<<<(Np + 255) / 256, 256, 0, stream>>>(b, dst, N, Np);
    };

    // 1) pack x -> bf16 [B,S,DP]
    pack_x<<<BB * SSEQ, 128, 0, stream>>>(seqs, ts, codes, tsc, emb, BF(o_xb));

    // 2) weight prep (transpose+pad to bf16), bias pad
    prep(faq_w, BF(o_faqT), DD, DD, DP, DP);
    prep(fak_w, BF(o_fakT), DD, DD, DP, DP);
    prep(fav_w, BF(o_favT), DD, DD, DP, DP);
    prep(taq_w, BF(o_taqT), SSEQ, SSEQ, SSEQ, SSEQ);
    prep(tak_w, BF(o_takT), SSEQ, SSEQ, SSEQ, SSEQ);
    prep(tav_w, BF(o_tavT), SSEQ, SSEQ, SSEQ, SSEQ);
    padb(faq_b, FP(o_bfaq), DD, DP); padb(fak_b, FP(o_bfak), DD, DP);
    padb(fav_b, FP(o_bfav), DD, DP);
    padb(taq_b, FP(o_btaq), SSEQ, SSEQ); padb(tak_b, FP(o_btak), SSEQ, SSEQ);
    padb(tav_b, FP(o_btav), SSEQ, SSEQ);

    const long long sBSDP = (long long)SSEQ * DP;

    // 3) feature-attention projections: q1, k1 [S,DP]; v1 stored transposed [DP,S]
    gemm(BF(o_xb), sBSDP, DP, BF(o_faqT), 0, DP, BF(o_q1),  sBSDP, DP, FP(o_bfaq), SSEQ, DP, DP, 0, 0);
    gemm(BF(o_xb), sBSDP, DP, BF(o_fakT), 0, DP, BF(o_k1),  sBSDP, DP, FP(o_bfak), SSEQ, DP, DP, 0, 0);
    gemm(BF(o_xb), sBSDP, DP, BF(o_favT), 0, DP, BF(o_v1t), sBSDP, SSEQ, FP(o_bfav), SSEQ, DP, DP, 0, 1);

    // 4) scores1 = q1 @ k1^T (BT = k1 directly), f32 [S,S]
    gemm(BF(o_q1), sBSDP, DP, BF(o_k1), sBSDP, DP, FP(o_s),
         (long long)SSEQ * SSEQ, SSEQ, nullptr, SSEQ, SSEQ, DP, 1, 0);

    // 5) softmax1: aw1 f32 -> d_out, bf16 probs -> p1
    softmax_rows<<<BB * SSEQ, 256, 0, stream>>>(FP(o_s), out_aw1, BF(o_p),
                                                SSEQ, SSEQ, SSEQ, SSEQ,
                                                1.f / sqrtf((float)DD));

    // 6) av1 = p1 @ v1 (BT = v1^T [DP,S]) -> bf16 [S,DP]  (reuse q1 buffer)
    gemm(BF(o_p), (long long)SSEQ * SSEQ, SSEQ, BF(o_v1t), sBSDP, SSEQ,
         BF(o_q1), sBSDP, DP, nullptr, SSEQ, DP, SSEQ, 0, 0);

    // 7) y = (av1 + PE)^T -> bf16 [DP,S] (reuse k1 buffer)
    build_y<<<BB * DP, 256, 0, stream>>>(BF(o_q1), BF(o_k1));

    // 8) temporal projections: q2 [DP,S] (reuse xb), k2 (reuse v1t), v2^T [S,DP] (reuse q1)
    gemm(BF(o_k1), sBSDP, SSEQ, BF(o_taqT), 0, SSEQ, BF(o_xb),  sBSDP, SSEQ, FP(o_btaq), DP, SSEQ, SSEQ, 0, 0);
    gemm(BF(o_k1), sBSDP, SSEQ, BF(o_takT), 0, SSEQ, BF(o_v1t), sBSDP, SSEQ, FP(o_btak), DP, SSEQ, SSEQ, 0, 0);
    gemm(BF(o_k1), sBSDP, SSEQ, BF(o_tavT), 0, SSEQ, BF(o_q1),  sBSDP, DP,   FP(o_btav), DP, SSEQ, SSEQ, 0, 1);

    // 9) scores2 = q2 @ k2^T (BT = k2), f32 [DP,DP]
    gemm(BF(o_xb), sBSDP, SSEQ, BF(o_v1t), sBSDP, SSEQ, FP(o_s),
         (long long)DP * DP, DP, nullptr, DP, DP, SSEQ, 1, 0);

    // 10) softmax2: aw2 f32 [401x401] -> d_out, padded bf16 probs -> p2
    softmax_rows<<<BB * DP, 256, 0, stream>>>(FP(o_s), out_aw2, BF(o_p),
                                              DP, DD, DP, DD,
                                              1.f / sqrtf((float)SSEQ));

    // 11) av2 = p2 @ v2 (BT = v2^T [S,DP]) -> f32 [DP,S] (reuse score buffer)
    gemm(BF(o_p), (long long)DP * DP, DP, BF(o_q1), sBSDP, DP,
         FP(o_s), sBSDP, SSEQ, nullptr, DP, SSEQ, DP, 1, 0);

    // 12) agg = mean_s av2 -> h[:, 0:DD]
    mean_rows<<<BB * DD, 256, 0, stream>>>(FP(o_s), FP(o_h));

    // 13) static attention -> aw3 (d_out) + h[:, DD:DD+64]
    static_attn<<<1, 256, 0, stream>>>(sv, saq_w, saq_b, sak_w, sak_b,
                                       sav_w, sav_b, out_aw3, FP(o_h));

    // 14) MLP head -> out[64]
    mlp_head<<<1, 256, 0, stream>>>(FP(o_h), fc1_w, fc1_b, fc2_w, fc2_b, out_main);
}
